// MyAttention_16320875725116
// MI455X (gfx1250) — matrix operand order repaired and checked by reading, not compiled, not run on hardware
//
#include <hip/hip_runtime.h>
#include <hip/hip_bf16.h>

// MI455X / gfx1250: wave32, WMMA 16x16x32 bf16, TDM tensor_load_to_lds.
typedef __attribute__((ext_vector_type(16))) __bf16 v16bf;
typedef __attribute__((ext_vector_type(8)))  __bf16 bf16x8;
typedef __attribute__((ext_vector_type(8)))  float  v8f;
typedef unsigned int u32x4 __attribute__((ext_vector_type(4)));
typedef int          i32x8 __attribute__((ext_vector_type(8)));
typedef int          i32x4 __attribute__((ext_vector_type(4)));

union Frag { v16bf v; bf16x8 h[2]; };

static __device__ __forceinline__ v8f wmma_bf16(v16bf a, v16bf b, v8f c) {
  // (neg_a, A, neg_b, B, c_mod, C, reuse_a, reuse_b)
  return __builtin_amdgcn_wmma_f32_16x16x32_bf16(false, a, false, b, (short)0, c,
                                                 false, false);
}

#define ZERO8 {0.f,0.f,0.f,0.f,0.f,0.f,0.f,0.f}

// ---------------------------------------------------------------------------
// TDM: DMA one 64-row x 128-f32 tile (rows strided 2048*512 floats in global)
// into LDS at lds_off, padding +4 DWORDs per 128-DWORD row (stride 132 f32).
// D# per CDNA5 ISA.
// ---------------------------------------------------------------------------
static __device__ __forceinline__ void tdm_load_tile64x128(const float* gsrc,
                                                           unsigned lds_off) {
  unsigned long long ga = (unsigned long long)(uintptr_t)gsrc;
  u32x4 g0 = { 1u,                                     // count=1, is_restore=0
               lds_off,                                // lds_addr
               (unsigned)(ga & 0xffffffffu),           // global_addr[31:0]
               (unsigned)((ga >> 32) & 0x01ffffffu) | (2u << 30) }; // [56:32]|type=2
  // data_size=2(4B), pad_enable, pad_interval=6(128 DW), pad_amount=3(4 DW)
  i32x8 g1 = { (int)0x07920000,
               (int)(128u << 16),       // tensor_dim0 = 128
               (int)(64u  << 16),       // tensor_dim1 = 64
               (int)(128u << 16),       // tile_dim0   = 128
               64,                      // tile_dim1   = 64, tile_dim2 = 0
               0x00100000,              // tensor_dim0_stride = 2048*512
               0, 0 };
  i32x4 gz = {0, 0, 0, 0};
#if __clang_major__ >= 23
  i32x8 gz8 = {0, 0, 0, 0, 0, 0, 0, 0};
  __builtin_amdgcn_tensor_load_to_lds(g0, g1, gz, gz, gz8, 0);
#else
  __builtin_amdgcn_tensor_load_to_lds(g0, g1, gz, gz, 0);
#endif
}

// Convert 16 f32 (A-fragment halves per ISA layout) from LDS row to bf16 frag.
static __device__ __forceinline__ v16bf a_frag_from_f32(const float* rowPtr, int hi) {
  const float4 f0 = *(const float4*)(rowPtr + hi * 8);
  const float4 f1 = *(const float4*)(rowPtr + hi * 8 + 4);
  const float4 f2 = *(const float4*)(rowPtr + 16 + hi * 8);
  const float4 f3 = *(const float4*)(rowPtr + 16 + hi * 8 + 4);
  v16bf r;
  r[0]  = (__bf16)f0.x; r[1]  = (__bf16)f0.y; r[2]  = (__bf16)f0.z; r[3]  = (__bf16)f0.w;
  r[4]  = (__bf16)f1.x; r[5]  = (__bf16)f1.y; r[6]  = (__bf16)f1.z; r[7]  = (__bf16)f1.w;
  r[8]  = (__bf16)f2.x; r[9]  = (__bf16)f2.y; r[10] = (__bf16)f2.z; r[11] = (__bf16)f2.w;
  r[12] = (__bf16)f3.x; r[13] = (__bf16)f3.y; r[14] = (__bf16)f3.z; r[15] = (__bf16)f3.w;
  return r;
}

// ---------------------------------------------------------------------------
// Generic bf16 WMMA GEMM: C[m,n] = sum_k A[m,k]*B[k,n] (+bias[n])
// A element at A[sar*m + sac*k + az*z], B at B[sbr*k + sbc*n + bz*z].
// 64x64 tile per 256-thread block (8 waves, each 16x32 of output).
// ---------------------------------------------------------------------------
template <bool WRITE_BF16>
__global__ __launch_bounds__(256) void gemm_bf16(
    const float* __restrict__ A, long long sar, long long sac, long long az,
    const float* __restrict__ B, long long sbr, long long sbc, long long bz,
    float* __restrict__ Cf, __bf16* __restrict__ Cb, long long ldc, long long cz,
    const float* __restrict__ bias, int Kdim)
{
  __shared__ __bf16 As[64 * 40];   // row-major [m][k], stride 40 (80B, 16B-aligned)
  __shared__ __bf16 Bt[64 * 40];   // N-major  [n][k]
  const int tid = threadIdx.x, lane = tid & 31, wave = tid >> 5;
  const int wm = wave >> 1, wn = wave & 1;
  const int hi = lane >> 4, lo = lane & 15;
  const long long z   = blockIdx.z;
  const long long gm0 = (long long)blockIdx.y * 64;
  const long long gn0 = (long long)blockIdx.x * 64;
  A += az * z; B += bz * z;

  v8f acc0 = ZERO8, acc1 = ZERO8;
  for (int k0 = 0; k0 < Kdim; k0 += 32) {
    for (int i = tid; i < 2048; i += 256) {
      int m = i >> 5, k = i & 31;
      As[m * 40 + k] = (__bf16)A[sar * (gm0 + m) + sac * (k0 + k)];
      Bt[m * 40 + k] = (__bf16)B[sbr * (k0 + k) + sbc * (gn0 + m)];
    }
    __syncthreads();
    Frag a, b0, b1;
    const __bf16* ap = As + (wm * 16 + lo) * 40;
    a.h[0] = *(const bf16x8*)(ap + hi * 8);
    a.h[1] = *(const bf16x8*)(ap + 16 + hi * 8);
    const __bf16* bp0 = Bt + (wn * 32 + lo) * 40;
    b0.h[0] = *(const bf16x8*)(bp0 + hi * 8);
    b0.h[1] = *(const bf16x8*)(bp0 + 16 + hi * 8);
    const __bf16* bp1 = Bt + (wn * 32 + 16 + lo) * 40;
    b1.h[0] = *(const bf16x8*)(bp1 + hi * 8);
    b1.h[1] = *(const bf16x8*)(bp1 + 16 + hi * 8);
    acc0 = wmma_bf16(a.v, b0.v, acc0);
    acc1 = wmma_bf16(a.v, b1.v, acc1);
    __syncthreads();
  }
  for (int nb = 0; nb < 2; ++nb) {
    v8f acc = nb ? acc1 : acc0;
    long long col = gn0 + wn * 32 + nb * 16 + lo;
    float bv = bias ? bias[col] : 0.f;
    for (int i = 0; i < 8; ++i) {
      long long row = gm0 + wm * 16 + i + 8 * hi;
      long long idx = row * ldc + cz * z + col;
      float val = acc[i] + bv;
      if (WRITE_BF16) Cb[idx] = (__bf16)val; else Cf[idx] = val;
    }
  }
}

// ---------------------------------------------------------------------------
// Fused attention core: one block per token n. TDM streams K/V as 8 chunks of
// 64x128 f32 into ping-pong LDS stages; WMMA for logits and p^T@V.
// ---------------------------------------------------------------------------
__global__ __launch_bounds__(256) void attn_core(
    const float* __restrict__ keys, const float* __restrict__ values,
    const int* __restrict__ mask, const __bf16* __restrict__ U,
    const float* __restrict__ bkq, float* __restrict__ W)
{
  __shared__ float  stage[2][64 * 132]; // TDM-padded f32 tiles (stride 132)
  __shared__ __bf16 vbt[128 * 64];      // V chunk, masked, [c_local][k]
  __shared__ __bf16 ubt[16 * 512];      // u^T: rows h (8 valid + 8 zero)
  __shared__ __bf16 pa[16 * 64];        // p^T: rows h (8 valid + 8 zero)
  __shared__ float  lgs[64 * 16];       // logits [k][h-pad16]
  __shared__ float  keepf[64];

  const int tid = threadIdx.x, lane = tid & 31, wave = tid >> 5;
  const int hi = lane >> 4, lo = lane & 15;
  const long long n = blockIdx.x;

  const unsigned so0 = (unsigned)(uintptr_t)(void*)&stage[0][0];
  const unsigned so1 = (unsigned)(uintptr_t)(void*)&stage[1][0];

  // Kick off the DMA pipeline (chunks 0,1 = K cols [0,128),[128,256)).
  if (wave == 0) {
    tdm_load_tile64x128(keys + n * 512,       so0);
    tdm_load_tile64x128(keys + n * 512 + 128, so1);
  }

  if (tid < 64) keepf[tid] = mask[n * 64 + tid] ? 0.f : 1.f;
  for (int i = tid; i < 512; i += 256) pa[512 + i] = (__bf16)0.f;  // rows 8..15
  {
    const __bf16* Ub = U + n * 4096;
    for (int i = tid; i < 4096; i += 256) { ubt[i] = Ub[i]; ubt[4096 + i] = (__bf16)0.f; }
  }

  v8f acc = ZERO8;  // logits accumulator (waves 0..3)

  for (int c = 0; c < 8; ++c) {
    if (wave == 0) {
      if (c == 7) __builtin_amdgcn_s_wait_tensorcnt(0);
      else        __builtin_amdgcn_s_wait_tensorcnt(1);
    }
    __syncthreads();
    const float* st = stage[c & 1];

    if (c < 4) {
      // ---- logits partial GEMM: A = K chunk (64x128), B = u^T ----
      // (no mask needed on K: masked rows are overwritten to -9e15 below)
      if (wave < 4) {
        const float* ar = st + (wave * 16 + lo) * 132;
        for (int k0 = 0; k0 < 128; k0 += 32) {
          v16bf a = a_frag_from_f32(ar + k0, hi);
          Frag b;
          const __bf16* br = ubt + lo * 512 + c * 128 + k0;
          b.h[0] = *(const bf16x8*)(br + hi * 8);
          b.h[1] = *(const bf16x8*)(br + 16 + hi * 8);
          acc = wmma_bf16(a, b.v, acc);
        }
      }
      if (c == 3) {
        if (wave < 4) {
          float bk = (lo < 8) ? bkq[n * 8 + lo] : 0.f;
          for (int i = 0; i < 8; ++i) {
            int kk = wave * 16 + i + 8 * hi;
            float val = (acc[i] + bk) * 0.125f * keepf[kk]; // ref: *scale then *keep
            if (val == 0.f) val = -9e15f;                    // ref: where(==0,-9e15)
            lgs[kk * 16 + lo] = val;
          }
        }
        __syncthreads();
        // softmax over k=64, one wave per head (V0 DMA already in flight)
        {
          float v1 = lgs[lane * 16 + wave], v2 = lgs[(lane + 32) * 16 + wave];
          float mx = fmaxf(v1, v2);
          for (int off = 16; off; off >>= 1) mx = fmaxf(mx, __shfl_xor(mx, off, 32));
          float e1 = __expf(v1 - mx), e2 = __expf(v2 - mx);
          float s = e1 + e2;
          for (int off = 16; off; off >>= 1) s += __shfl_xor(s, off, 32);
          float inv = 1.f / s;
          pa[wave * 64 + lane]      = (__bf16)(e1 * inv);
          pa[wave * 64 + lane + 32] = (__bf16)(e2 * inv);
        }
      }
    } else {
      // ---- V chunk: mask+convert to vbt[c_local][k], then W tile GEMM ----
      for (int i = tid; i < 2048; i += 256) {
        int k = i >> 5, c4 = i & 31;
        const float4 vv = *(const float4*)(st + k * 132 + c4 * 4);
        float kp = keepf[k];
        vbt[(c4 * 4 + 0) * 64 + k] = (__bf16)(vv.x * kp);
        vbt[(c4 * 4 + 1) * 64 + k] = (__bf16)(vv.y * kp);
        vbt[(c4 * 4 + 2) * 64 + k] = (__bf16)(vv.z * kp);
        vbt[(c4 * 4 + 3) * 64 + k] = (__bf16)(vv.w * kp);
      }
      __syncthreads();
      v8f acc2 = ZERO8;
      for (int ks = 0; ks < 2; ++ks) {
        Frag a, b;
        const __bf16* pr = pa + lo * 64 + ks * 32;
        a.h[0] = *(const bf16x8*)(pr + hi * 8);
        a.h[1] = *(const bf16x8*)(pr + 16 + hi * 8);
        const __bf16* br = vbt + (wave * 16 + lo) * 64 + ks * 32;
        b.h[0] = *(const bf16x8*)(br + hi * 8);
        b.h[1] = *(const bf16x8*)(br + 16 + hi * 8);
        acc2 = wmma_bf16(a.v, b.v, acc2);
      }
      const int colBase = (c - 4) * 128 + wave * 16;
      for (int i = 0; i < 8; ++i) {
        int hh = i + 8 * hi;
        if (hh < 8)
          W[n * 4096 + (long long)hh * 512 + colBase + lo] = acc2[i];
      }
    }
    __syncthreads();
    // Issue the DMA for chunk c+2 into the stage we just consumed.
    if (wave == 0 && c + 2 < 8) {
      int nc = c + 2;
      const float* src = (nc < 4)
          ? keys   + n * 512 + (long long)nc * 128
          : values + n * 512 + (long long)(nc - 4) * 128;
      tdm_load_tile64x128(src, (nc & 1) ? so1 : so0);
    }
  }
}

// ---------------------------------------------------------------------------
// Tiny precompute kernels
// ---------------------------------------------------------------------------
__global__ void m2_kernel(const float* __restrict__ WV, const float* __restrict__ WO,
                          float* __restrict__ M2)
{
  int idx = blockIdx.x * 256 + threadIdx.x;      // 262144 = 4096*64
  int f = idx >> 6, m = idx & 63;
  int h = f >> 9, c = f & 511;
  float s = 0.f;
  for (int j = 0; j < 64; ++j)
    s += WV[c * 512 + h * 64 + j] * WO[(h * 64 + j) * 64 + m];
  M2[idx] = s;
}

__global__ void c0_kernel(const float* __restrict__ bV, const float* __restrict__ bO,
                          const float* __restrict__ WO, float* __restrict__ c0)
{
  int m = threadIdx.x;                            // 64 threads
  float s = bO[m];
  for (int f = 0; f < 512; ++f) s += bV[f] * WO[f * 64 + m];
  c0[m] = s;
}

__global__ void bkq_kernel(const float* __restrict__ QH, const float* __restrict__ bK,
                           float* __restrict__ bkq)
{
  int idx = blockIdx.x * 256 + threadIdx.x;       // 16384 = 2048*8
  int nn = idx >> 3, h = idx & 7;
  const float* q = QH + (long long)nn * 512 + h * 64;
  const float* b = bK + h * 64;
  float s = 0.f;
  for (int j = 0; j < 64; ++j) s += b[j] * q[j];
  bkq[idx] = s;
}

// ---------------------------------------------------------------------------
extern "C" void kernel_launch(void* const* d_in, const int* in_sizes, int n_in,
                              void* d_out, int out_size, void* d_ws, size_t ws_size,
                              hipStream_t stream)
{
  const float* queries = (const float*)d_in[0];
  const float* keys    = (const float*)d_in[1];
  const float* values  = (const float*)d_in[2];
  const int*   mask    = (const int*)  d_in[3];
  const float* WQ = (const float*)d_in[4];
  const float* bQ = (const float*)d_in[5];
  const float* WK = (const float*)d_in[6];
  const float* bK = (const float*)d_in[7];
  const float* WV = (const float*)d_in[8];
  const float* bV = (const float*)d_in[9];
  const float* WO = (const float*)d_in[10];
  const float* bO = (const float*)d_in[11];
  float* out = (float*)d_out;

  char* ws = (char*)d_ws;
  float*  QH  = (float*)(ws + 0);                        //  4 MB  (2048x512 f32)
  __bf16* U   = (__bf16*)(ws + (4ll  << 20));            // 16 MB  (2048x8x512 bf16)
  float*  bkq = (float*)(ws + (20ll << 20));             // 64 KB  (2048x8 f32)
  float*  W   = (float*)(ws + (21ll << 20));             // 32 MB  (2048x4096 f32)
  float*  M2  = (float*)(ws + (55ll << 20));             //  1 MB  (4096x64 f32)
  float*  c0  = (float*)(ws + (56ll << 20));             // 256 B

  // Fold WV/WO into M2 and the biases into c0 (tiny, independent).
  m2_kernel<<<1024, 256, 0, stream>>>(WV, WO, M2);
  c0_kernel<<<1, 64, 0, stream>>>(bV, bO, WO, c0);

  // QH = Q @ WQ + bQ               (2048x512x512)
  gemm_bf16<false><<<dim3(8, 32, 1), 256, 0, stream>>>(
      queries, 512, 1, 0,  WQ, 512, 1, 0,  QH, nullptr, 512, 0,  bQ, 512);

  // bkq[n,h] = bK_h . qh_h
  bkq_kernel<<<64, 256, 0, stream>>>(QH, bK, bkq);

  // U[n,h,:] = qh_h @ WK_h^T       (8 heads via grid.z; 2048x512, K=64)
  gemm_bf16<true><<<dim3(8, 32, 8), 256, 0, stream>>>(
      QH, 512, 1, 64,  WK, 1, 512, 64,  nullptr, U, 4096, 512,  nullptr, 64);

  // Fused streaming attention core -> W[n,h,c]
  attn_core<<<2048, 256, 0, stream>>>(keys, values, mask, U, bkq, W);

  // out = W @ M2 + c0              (2048x4096x64)
  gemm_bf16<false><<<dim3(1, 32, 1), 256, 0, stream>>>(
      W, 4096, 1, 0,  M2, 64, 1, 0,  out, nullptr, 64, 0,  c0, 4096);
}
